// AttentionBlock_42709154791463
// MI455X (gfx1250) — matrix operand order repaired and checked by reading, not compiled, not run on hardware
//
#include <hip/hip_runtime.h>
#include <hip/hip_bf16.h>
#include <math.h>

typedef __attribute__((ext_vector_type(16))) _Float16 v16h;
typedef __attribute__((ext_vector_type(8)))  float    v8f;

#define BQ   2
#define SEQ  2048
#define ED   1024
#define NH   16
#define HD   64
#define HID  4096
#define MROWS (BQ*SEQ)          // 4096 token rows

union FragH { v16h h; uint4 u[2]; };

// ---- WMMA fragment loaders (CDNA5 16-bit layouts, wave32) -----------------
// A operand 16x32 f16: lane l -> row (l&15); K pattern: [kb..kb+7],[kb+16..kb+23], kb=(l>>4)*8
__device__ inline v16h load_a_frag(const _Float16* p, int ld) {
  int l = threadIdx.x & 31;
  const _Float16* q = p + (size_t)(l & 15) * ld + ((l >> 4) << 3);
  FragH f;
  f.u[0] = *(const uint4*)(q);
  f.u[1] = *(const uint4*)(q + 16);
  return f.h;
}
// B operand 32x16 f16 (loaded as rows of B^T): lane l -> col (l&15); K pattern: [kb..kb+7],[kb+16..kb+23], kb=(l>>4)*8
__device__ inline v16h load_b_frag(const _Float16* p, int ld) {
  int l = threadIdx.x & 31;
  const _Float16* q = p + (size_t)(l & 15) * ld + ((l >> 4) << 3);
  FragH f;
  f.u[0] = *(const uint4*)(q);
  f.u[1] = *(const uint4*)(q + 16);
  return f.h;
}
__device__ inline v8f wmma_f16(v16h a, v16h b, v8f c) {
  return __builtin_amdgcn_wmma_f32_16x16x32_f16(false, a, false, b, (short)0, c, false, false);
}

// ---- fp32 -> f16 weight convert -------------------------------------------
__global__ void cvt_f32_f16(const float* __restrict__ src, _Float16* __restrict__ dst, int n) {
  int i = blockIdx.x * blockDim.x + threadIdx.x;
  if (i < n) dst[i] = (_Float16)src[i];
}

// ---- LayerNorm (wave per row, 1024 cols) ----------------------------------
__global__ __launch_bounds__(256)
void layernorm_f16(const float* __restrict__ X, const float* __restrict__ gamma,
                   const float* __restrict__ beta, _Float16* __restrict__ Y) {
  int wave = threadIdx.x >> 5, lane = threadIdx.x & 31;
  int row  = blockIdx.x * 8 + wave;
  const float* x = X + (size_t)row * ED;
  float vals[32], s = 0.f, sq = 0.f;
#pragma unroll
  for (int i = 0; i < 32; ++i) {
    float v = x[lane + i * 32];
    vals[i] = v; s += v; sq += v * v;
  }
  for (int m = 1; m < 32; m <<= 1) { s += __shfl_xor(s, m, 32); sq += __shfl_xor(sq, m, 32); }
  float mean = s * (1.f / ED);
  float var  = sq * (1.f / ED) - mean * mean;
  float rstd = rsqrtf(var + 1e-5f);
  _Float16* y = Y + (size_t)row * ED;
#pragma unroll
  for (int i = 0; i < 32; ++i) {
    int c = lane + i * 32;
    y[c] = (_Float16)((vals[i] - mean) * rstd * gamma[c] + beta[c]);
  }
}

// ---- GEMM: C(f16) = A(f16)[M,K] * W(f16)[NC,K]^T   (wave -> 32x64 tile) ---
__global__ __launch_bounds__(32)
void gemm_f16out(const _Float16* __restrict__ A, const _Float16* __restrict__ Bw,
                 _Float16* __restrict__ C, int K, int NC) {
  int mbase = blockIdx.x * 32, nbase = blockIdx.y * 64;
  v8f acc[2][4] = {};
  const _Float16* a0p = A + (size_t)mbase * K;
  const _Float16* a1p = a0p + (size_t)16 * K;
  const _Float16* bp  = Bw + (size_t)nbase * K;
  for (int k = 0; k < K; k += 32) {
    __builtin_prefetch(a0p + k + 256, 0, 3);
    __builtin_prefetch(bp  + k + 256, 0, 3);
    v16h a0 = load_a_frag(a0p + k, K);
    v16h a1 = load_a_frag(a1p + k, K);
#pragma unroll
    for (int j = 0; j < 4; ++j) {
      v16h b = load_b_frag(bp + (size_t)(16 * j) * K + k, K);
      acc[0][j] = wmma_f16(a0, b, acc[0][j]);
      acc[1][j] = wmma_f16(a1, b, acc[1][j]);
    }
  }
  int l = threadIdx.x & 31, n = l & 15, mo = (l >> 4) * 8;
#pragma unroll
  for (int v = 0; v < 8; ++v) {
    size_t m0 = mbase + mo + v, m1 = m0 + 16;
#pragma unroll
    for (int j = 0; j < 4; ++j) {
      C[m0 * NC + nbase + 16 * j + n] = (_Float16)acc[0][j][v];
      C[m1 * NC + nbase + 16 * j + n] = (_Float16)acc[1][j][v];
    }
  }
}

// ---- GEMM: C(f32) = A*W^T + bias + resid   (wave -> 32x64 tile) -----------
__global__ __launch_bounds__(32)
void gemm_bias_res_f32(const _Float16* __restrict__ A, const _Float16* __restrict__ Bw,
                       const float* __restrict__ bias, const float* __restrict__ resid,
                       float* __restrict__ C, int K, int NC) {
  int mbase = blockIdx.x * 32, nbase = blockIdx.y * 64;
  v8f acc[2][4] = {};
  const _Float16* a0p = A + (size_t)mbase * K;
  const _Float16* a1p = a0p + (size_t)16 * K;
  const _Float16* bp  = Bw + (size_t)nbase * K;
  for (int k = 0; k < K; k += 32) {
    __builtin_prefetch(a0p + k + 256, 0, 3);
    __builtin_prefetch(bp  + k + 256, 0, 3);
    v16h a0 = load_a_frag(a0p + k, K);
    v16h a1 = load_a_frag(a1p + k, K);
#pragma unroll
    for (int j = 0; j < 4; ++j) {
      v16h b = load_b_frag(bp + (size_t)(16 * j) * K + k, K);
      acc[0][j] = wmma_f16(a0, b, acc[0][j]);
      acc[1][j] = wmma_f16(a1, b, acc[1][j]);
    }
  }
  int l = threadIdx.x & 31, n = l & 15, mo = (l >> 4) * 8;
  float bb[4];
#pragma unroll
  for (int j = 0; j < 4; ++j) bb[j] = bias[nbase + 16 * j + n];
#pragma unroll
  for (int v = 0; v < 8; ++v) {
    size_t m0 = mbase + mo + v, m1 = m0 + 16;
#pragma unroll
    for (int j = 0; j < 4; ++j) {
      size_t c = nbase + 16 * j + n;
      C[m0 * NC + c] = acc[0][j][v] + bb[j] + resid[m0 * NC + c];
      C[m1 * NC + c] = acc[1][j][v] + bb[j] + resid[m1 * NC + c];
    }
  }
}

// ---- Fused FC1 + gated GELU: ff = gelu(x*Wg^T+bg) * (x*Wv^T+bv) -----------
__global__ __launch_bounds__(32)
void gemm_glu(const _Float16* __restrict__ A, const _Float16* __restrict__ Wg,
              const _Float16* __restrict__ Wv, const float* __restrict__ bias,
              _Float16* __restrict__ C, int K) {
  int mbase = blockIdx.x * 32, nbase = blockIdx.y * 32;
  v8f g00 = {}, g01 = {}, g10 = {}, g11 = {};
  v8f w00 = {}, w01 = {}, w10 = {}, w11 = {};
  const _Float16* a0p = A  + (size_t)mbase * K;
  const _Float16* a1p = a0p + (size_t)16 * K;
  const _Float16* bg0 = Wg + (size_t)nbase * K;
  const _Float16* bg1 = bg0 + (size_t)16 * K;
  const _Float16* bv0 = Wv + (size_t)nbase * K;
  const _Float16* bv1 = bv0 + (size_t)16 * K;
  for (int k = 0; k < K; k += 32) {
    __builtin_prefetch(bg0 + k + 256, 0, 3);
    __builtin_prefetch(bv0 + k + 256, 0, 3);
    v16h a0 = load_a_frag(a0p + k, K);
    v16h a1 = load_a_frag(a1p + k, K);
    v16h b0 = load_b_frag(bg0 + k, K);
    v16h b1 = load_b_frag(bg1 + k, K);
    g00 = wmma_f16(a0, b0, g00); g01 = wmma_f16(a0, b1, g01);
    g10 = wmma_f16(a1, b0, g10); g11 = wmma_f16(a1, b1, g11);
    b0 = load_b_frag(bv0 + k, K);
    b1 = load_b_frag(bv1 + k, K);
    w00 = wmma_f16(a0, b0, w00); w01 = wmma_f16(a0, b1, w01);
    w10 = wmma_f16(a1, b0, w10); w11 = wmma_f16(a1, b1, w11);
  }
  int l = threadIdx.x & 31, n = l & 15, mo = (l >> 4) * 8;
  int c0 = nbase + n, c1 = nbase + 16 + n;
  float bg_0 = bias[c0], bg_1 = bias[c1];
  float bv_0 = bias[HID + c0], bv_1 = bias[HID + c1];
#pragma unroll
  for (int v = 0; v < 8; ++v) {
    size_t m0 = mbase + mo + v, m1 = m0 + 16;
    float g, w;
    g = g00[v] + bg_0; w = w00[v] + bv_0;
    C[m0 * HID + c0] = (_Float16)(0.5f * g * (1.f + erff(g * 0.70710678f)) * w);
    g = g01[v] + bg_1; w = w01[v] + bv_1;
    C[m0 * HID + c1] = (_Float16)(0.5f * g * (1.f + erff(g * 0.70710678f)) * w);
    g = g10[v] + bg_0; w = w10[v] + bv_0;
    C[m1 * HID + c0] = (_Float16)(0.5f * g * (1.f + erff(g * 0.70710678f)) * w);
    g = g11[v] + bg_1; w = w11[v] + bv_1;
    C[m1 * HID + c1] = (_Float16)(0.5f * g * (1.f + erff(g * 0.70710678f)) * w);
  }
}

// ---- RoPE + head rearrange (q scaled by 1/sqrt(HD), V stored transposed) --
__global__ __launch_bounds__(256)
void rope_rearrange(const _Float16* __restrict__ qkv, _Float16* __restrict__ Qh,
                    _Float16* __restrict__ Kh, _Float16* __restrict__ Vt) {
  int idx = blockIdx.x * blockDim.x + threadIdx.x;  // over B*SEQ*NH*(HD/2)
  int i    = idx & 31;
  int h    = (idx >> 5) & (NH - 1);
  int npos = (idx >> 9) & (SEQ - 1);
  int b    = idx >> 20;
  float theta = __powf(10000.f, -2.f * (float)i / (float)HD);
  float sn, cs;
  __sincosf((float)npos * theta, &sn, &cs);
  size_t row = (size_t)b * SEQ + npos;
  const _Float16* base = qkv + row * (3 * ED) + h * HD + 2 * i;
  float qr = (float)base[0],        qi = (float)base[1];
  float kr = (float)base[ED],       ki = (float)base[ED + 1];
  float v0 = (float)base[2 * ED],   v1 = (float)base[2 * ED + 1];
  size_t bh = (size_t)(b * NH + h);
  _Float16* qo = Qh + (bh * SEQ + npos) * HD + 2 * i;
  qo[0] = (_Float16)((qr * cs - qi * sn) * 0.125f);
  qo[1] = (_Float16)((qr * sn + qi * cs) * 0.125f);
  _Float16* ko = Kh + (bh * SEQ + npos) * HD + 2 * i;
  ko[0] = (_Float16)(kr * cs - ki * sn);
  ko[1] = (_Float16)(kr * sn + ki * cs);
  Vt[(bh * HD + 2 * i) * SEQ + npos]     = (_Float16)v0;
  Vt[(bh * HD + 2 * i + 1) * SEQ + npos] = (_Float16)v1;
}

// ---- Flash attention: wave per (b,h,16 query rows), online softmax --------
__global__ __launch_bounds__(32)
void flash_attn(const _Float16* __restrict__ Qh, const _Float16* __restrict__ Kh,
                const _Float16* __restrict__ Vt, _Float16* __restrict__ O) {
  int bh = blockIdx.x;       // b*NH + h
  int qt = blockIdx.y;       // query tile
  const _Float16* qbase = Qh + ((size_t)bh * SEQ + qt * 16) * HD;
  v16h qa0 = load_a_frag(qbase, HD);        // d = 0..31
  v16h qa1 = load_a_frag(qbase + 32, HD);   // d = 32..63

  v8f o0 = {}, o1 = {}, o2 = {}, o3 = {};
  float mrow[8], lrow[8];
#pragma unroll
  for (int v = 0; v < 8; ++v) { mrow[v] = -1e30f; lrow[v] = 0.f; }

  __shared__ __align__(16) _Float16 Pt[16 * 32];
  int l = threadIdx.x & 31, nlane = l & 15, half = l >> 4;

  for (int kc = 0; kc < SEQ; kc += 32) {
    const _Float16* kb = Kh + ((size_t)bh * SEQ + kc) * HD;
    v8f s0 = {}, s1 = {};
    s0 = wmma_f16(qa0, load_b_frag(kb, HD), s0);
    s0 = wmma_f16(qa1, load_b_frag(kb + 32, HD), s0);
    s1 = wmma_f16(qa0, load_b_frag(kb + 16 * HD, HD), s1);
    s1 = wmma_f16(qa1, load_b_frag(kb + 16 * HD + 32, HD), s1);

#pragma unroll
    for (int v = 0; v < 8; ++v) {
      float cm = fmaxf(s0[v], s1[v]);                 // keys live on 16 lanes/half
      for (int m = 1; m < 16; m <<= 1) cm = fmaxf(cm, __shfl_xor(cm, m, 32));
      float mnew  = fmaxf(mrow[v], cm);
      float alpha = __expf(mrow[v] - mnew);
      mrow[v] = mnew;
      float p0 = __expf(s0[v] - mnew);
      float p1 = __expf(s1[v] - mnew);
      float rs = p0 + p1;
      for (int m = 1; m < 16; m <<= 1) rs += __shfl_xor(rs, m, 32);
      lrow[v] = lrow[v] * alpha + rs;
      o0[v] *= alpha; o1[v] *= alpha; o2[v] *= alpha; o3[v] *= alpha;
      int m = half * 8 + v;                            // C-layout row
      Pt[m * 32 + nlane]      = (_Float16)p0;
      Pt[m * 32 + 16 + nlane] = (_Float16)p1;
    }
    asm volatile("" ::: "memory");                     // P stores before A-frag reload
    v16h pa = load_a_frag(Pt, 32);                     // P as A operand (16x32)
    const _Float16* vb = Vt + (size_t)bh * HD * SEQ + kc;
    o0 = wmma_f16(pa, load_b_frag(vb,            SEQ), o0);
    o1 = wmma_f16(pa, load_b_frag(vb + 16 * SEQ, SEQ), o1);
    o2 = wmma_f16(pa, load_b_frag(vb + 32 * SEQ, SEQ), o2);
    o3 = wmma_f16(pa, load_b_frag(vb + 48 * SEQ, SEQ), o3);
    asm volatile("" ::: "memory");
  }

  int b = bh >> 4, h = bh & 15;
  size_t rowbase = (size_t)b * SEQ + qt * 16;
#pragma unroll
  for (int v = 0; v < 8; ++v) {
    int m = half * 8 + v;
    float inv = 1.f / lrow[v];
    size_t r = (rowbase + m) * ED + h * HD + nlane;
    O[r]      = (_Float16)(o0[v] * inv);
    O[r + 16] = (_Float16)(o1[v] * inv);
    O[r + 32] = (_Float16)(o2[v] * inv);
    O[r + 48] = (_Float16)(o3[v] * inv);
  }
}

// ---- launch ---------------------------------------------------------------
extern "C" void kernel_launch(void* const* d_in, const int* in_sizes, int n_in,
                              void* d_out, int out_size, void* d_ws, size_t ws_size,
                              hipStream_t stream) {
  const float* x      = (const float*)d_in[0];
  const float* w_qkv  = (const float*)d_in[1];
  const float* w_out  = (const float*)d_in[2];
  const float* b_out  = (const float*)d_in[3];
  const float* gamma1 = (const float*)d_in[4];
  const float* beta1  = (const float*)d_in[5];
  const float* gamma2 = (const float*)d_in[6];
  const float* beta2  = (const float*)d_in[7];
  const float* w_fc1  = (const float*)d_in[8];
  const float* b_fc1  = (const float*)d_in[9];
  const float* w_fc2  = (const float*)d_in[10];
  const float* b_fc2  = (const float*)d_in[11];

  size_t off = 0;
  auto alloc = [&](size_t bytes) {
    void* p = (char*)d_ws + off;
    off += (bytes + 255) & ~(size_t)255;
    return p;
  };
  _Float16* xln   = (_Float16*)alloc((size_t)MROWS * ED * 2);
  _Float16* wqkvh = (_Float16*)alloc((size_t)3 * ED * ED * 2);
  _Float16* qkvh  = (_Float16*)alloc((size_t)MROWS * 3 * ED * 2);
  _Float16* qh    = (_Float16*)alloc((size_t)MROWS * ED * 2);
  _Float16* kh    = (_Float16*)alloc((size_t)MROWS * ED * 2);
  _Float16* vth   = (_Float16*)alloc((size_t)MROWS * ED * 2);
  _Float16* attnh = (_Float16*)alloc((size_t)MROWS * ED * 2);
  _Float16* wouth = (_Float16*)alloc((size_t)ED * ED * 2);
  float*    x1    = (float*)   alloc((size_t)MROWS * ED * 4);
  _Float16* x1ln  = (_Float16*)alloc((size_t)MROWS * ED * 2);
  _Float16* wfc1h = (_Float16*)alloc((size_t)2 * HID * ED * 2);
  _Float16* ffh   = (_Float16*)alloc((size_t)MROWS * HID * 2);
  _Float16* wfc2h = (_Float16*)alloc((size_t)ED * HID * 2);

  // weight conversion (fp32 -> f16)
  cvt_f32_f16<<<(3 * ED * ED) / 256, 256, 0, stream>>>(w_qkv, wqkvh, 3 * ED * ED);
  cvt_f32_f16<<<(ED * ED) / 256, 256, 0, stream>>>(w_out, wouth, ED * ED);
  cvt_f32_f16<<<(2 * HID * ED) / 256, 256, 0, stream>>>(w_fc1, wfc1h, 2 * HID * ED);
  cvt_f32_f16<<<(ED * HID) / 256, 256, 0, stream>>>(w_fc2, wfc2h, ED * HID);

  // LN1 -> QKV -> RoPE -> attention
  layernorm_f16<<<MROWS / 8, 256, 0, stream>>>(x, gamma1, beta1, xln);
  gemm_f16out<<<dim3(MROWS / 32, (3 * ED) / 64), 32, 0, stream>>>(xln, wqkvh, qkvh, ED, 3 * ED);
  rope_rearrange<<<(BQ * SEQ * NH * (HD / 2)) / 256, 256, 0, stream>>>(qkvh, qh, kh, vth);
  flash_attn<<<dim3(BQ * NH, SEQ / 16), 32, 0, stream>>>(qh, kh, vth, attnh);

  // out-proj + residual -> LN2 -> FC1+GLU -> FC2 + residual
  gemm_bias_res_f32<<<dim3(MROWS / 32, ED / 64), 32, 0, stream>>>(attnh, wouth, b_out, x, x1, ED, ED);
  layernorm_f16<<<MROWS / 8, 256, 0, stream>>>(x1, gamma2, beta2, x1ln);
  gemm_glu<<<dim3(MROWS / 32, HID / 32), 32, 0, stream>>>(x1ln, wfc1h, wfc1h + (size_t)HID * ED,
                                                          b_fc1, ffh, ED);
  gemm_bias_res_f32<<<dim3(MROWS / 32, ED / 64), 32, 0, stream>>>(ffh, wfc2h, b_fc2, x1,
                                                                  (float*)d_out, HID, ED);
}
